// MatGCN_36945308680387
// MI455X (gfx1250) — compile-verified, matrix operand order checked by reading, not run on hardware
//
#include <hip/hip_runtime.h>
#include <stdint.h>

#define D 128
#define NLAYERS 4
#define LN_EPS 1e-5f
#define MT 4   // M-subtiles per wave in the WMMA GEMM

typedef __attribute__((ext_vector_type(16))) __bf16 v16bf;
typedef __attribute__((ext_vector_type(8)))  float  v8f;

// ---------------- utility ----------------
__global__ __launch_bounds__(256) void k_zero(float* __restrict__ p, long long n) {
  long long i = (long long)blockIdx.x * blockDim.x + threadIdx.x;
  if (i < n) p[i] = 0.0f;
}

// degree over dst (self-loop +1 folded into k_dis)
__global__ __launch_bounds__(256) void k_deg(const long long* __restrict__ dst,
                                             float* __restrict__ deg, long long e) {
  long long i = (long long)blockIdx.x * blockDim.x + threadIdx.x;
  if (i < e) atomicAdd(&deg[dst[i]], 1.0f);
}

__global__ __launch_bounds__(256) void k_dis(const float* __restrict__ deg,
                                             float* __restrict__ dis, int n) {
  int i = blockIdx.x * blockDim.x + threadIdx.x;
  if (i < n) dis[i] = rsqrtf(deg[i] + 1.0f);
}

// Wt[l][n][k] = (bf16) W[l][k][n]  -- transposed so B fragments load contiguously in k
__global__ __launch_bounds__(256) void k_prepw(const float* __restrict__ W,
                                               __bf16* __restrict__ Wt) {
  int i = blockIdx.x * 256 + threadIdx.x;          // 4*128*128 threads
  int l = i >> 14, r = i & 16383, nrow = r >> 7, k = r & 127;
  Wt[(l << 14) + (nrow << 7) + k] = (__bf16)W[(l << 14) + (k << 7) + nrow];
}

__global__ __launch_bounds__(256) void k_cvt(const float* __restrict__ x,
                                             __bf16* __restrict__ xb, long long n) {
  long long i = (long long)blockIdx.x * blockDim.x + threadIdx.x;
  if (i < n) xb[i] = (__bf16)x[i];
}

// ---------------- WMMA GEMM: H[N,128] = A[N,128] @ W[128,128] (bf16 in, f32 acc) ----
// 8 waves/block cover the 8 N-tiles; each wave preloads its B fragments once and
// sweeps MT M-subtiles (16 WMMAs / wave). grid.x = ceil(Mtiles / MT).
__global__ __launch_bounds__(256) void k_gemm(const __bf16* __restrict__ A,
                                              const __bf16* __restrict__ Bt, // [n][k]
                                              float* __restrict__ H,
                                              int n_nodes, int mtiles) {
  const int wid  = threadIdx.x >> 5;
  const int lane = threadIdx.x & 31;
  const int half = lane >> 4;          // lanes 0-15 vs 16-31
  const int mr   = lane & 15;

  // Preload B fragments for this wave's ntile (= wid), all 4 K-chunks.
  // 16-bit layout: VGPR0..3 <- K = kb+8*half+0..7 ; VGPR4..7 <- K = kb+16+8*half+0..7
  const __bf16* bp = Bt + (size_t)(wid * 16 + mr) * D;
  v16bf fb[4];
#pragma unroll
  for (int kc = 0; kc < 4; ++kc) {
    union { v16bf v; uint4 q[2]; } f;
    f.q[0] = *reinterpret_cast<const uint4*>(bp + kc * 32 + 8 * half);
    f.q[1] = *reinterpret_cast<const uint4*>(bp + kc * 32 + 16 + 8 * half);
    fb[kc] = f.v;
  }

  // Prefetch next block's first A rows (global_prefetch_b8).
  {
    long long pm = (long long)(blockIdx.x + 1) * MT * 16 + mr;
    if (pm < n_nodes) __builtin_prefetch(A + (size_t)pm * D, 0, 0);
  }

#pragma unroll
  for (int sm = 0; sm < MT; ++sm) {
    int mt = blockIdx.x * MT + sm;
    if (mt >= mtiles) break;                       // uniform branch, EXEC unchanged
    int arow = mt * 16 + mr;
    if (arow > n_nodes - 1) arow = n_nodes - 1;    // per-lane clamp, no EXEC change
    const __bf16* ap = A + (size_t)arow * D;

    v8f c = {};
#pragma unroll
    for (int kc = 0; kc < 4; ++kc) {
      union { v16bf v; uint4 q[2]; } fa;
      fa.q[0] = *reinterpret_cast<const uint4*>(ap + kc * 32 + 8 * half);
      fa.q[1] = *reinterpret_cast<const uint4*>(ap + kc * 32 + 16 + 8 * half);
      c = __builtin_amdgcn_wmma_f32_16x16x32_bf16(false, fa.v, false, fb[kc],
                                                  (short)0, c, false, false);
    }

    // D layout: VGPR r holds M = r + 8*half, N = lane&15
    int mbase = mt * 16 + 8 * half;
    float* outp = H + (size_t)(mt * 16 + 8 * half) * D + wid * 16 + mr;
    if (mt * 16 + 16 <= n_nodes) {                 // uniform fast path: clean stores
#pragma unroll
      for (int r = 0; r < 8; ++r) outp[(size_t)r * D] = c[r];
    } else {
#pragma unroll
      for (int r = 0; r < 8; ++r)
        if (mbase + r < n_nodes) outp[(size_t)r * D] = c[r];
    }
  }
}

// ---------------- agg = H * dis^2 (self loop), then edges scatter-add -------------
__global__ __launch_bounds__(256) void k_agg_init(const float* __restrict__ H,
                                                  const float* __restrict__ dis,
                                                  float* __restrict__ agg, int n) {
  long long gid = (long long)blockIdx.x * blockDim.x + threadIdx.x;
  int node = (int)(gid >> 5);
  if (node >= n) return;
  int q = (int)(gid & 31) << 2;
  float s = dis[node]; s *= s;
  float4 h4 = *reinterpret_cast<const float4*>(H + (size_t)node * D + q);
  float4 o  = make_float4(h4.x * s, h4.y * s, h4.z * s, h4.w * s);
  *reinterpret_cast<float4*>(agg + (size_t)node * D + q) = o;
}

__global__ __launch_bounds__(256) void k_scatter(const float* __restrict__ H,
                                                 const long long* __restrict__ src,
                                                 const long long* __restrict__ dst,
                                                 const float* __restrict__ dis,
                                                 float* __restrict__ agg, long long e) {
  long long gid = (long long)blockIdx.x * blockDim.x + threadIdx.x;
  long long ed = gid >> 5;
  if (ed >= e) return;
  int q = (int)(gid & 31) << 2;
  long long s = src[ed], t = dst[ed];
  float c = dis[s] * dis[t];
  float4 h4 = *reinterpret_cast<const float4*>(H + s * D + q);
  float* p = agg + t * D + q;
  atomicAdd(p + 0, h4.x * c); atomicAdd(p + 1, h4.y * c);
  atomicAdd(p + 2, h4.z * c); atomicAdd(p + 3, h4.w * c);
}

// ------------- fused: +bias, LayerNorm, (+residual), (ReLU); emits f32 + bf16 -------
__global__ __launch_bounds__(256) void k_finalize(const float* __restrict__ agg,
                                                  const float* __restrict__ bias,
                                                  const float* __restrict__ gamma,
                                                  const float* __restrict__ beta,
                                                  float* __restrict__ xf,
                                                  __bf16* __restrict__ xb,
                                                  int layer, int n) {
  int wid = threadIdx.x >> 5, lane = threadIdx.x & 31;
  int node = blockIdx.x * 8 + wid;
  if (node >= n) return;
  int d = lane << 2;
  float4 v  = *reinterpret_cast<const float4*>(agg + (size_t)node * D + d);
  float4 b4 = *reinterpret_cast<const float4*>(bias + d);
  v.x += b4.x; v.y += b4.y; v.z += b4.z; v.w += b4.w;
  float s = v.x + v.y + v.z + v.w;
#pragma unroll
  for (int off = 16; off > 0; off >>= 1) s += __shfl_xor(s, off, 32);
  float mu = s * (1.0f / 128.0f);
  float ax = v.x - mu, ay = v.y - mu, az = v.z - mu, aw = v.w - mu;
  float qs = ax * ax + ay * ay + az * az + aw * aw;
#pragma unroll
  for (int off = 16; off > 0; off >>= 1) qs += __shfl_xor(qs, off, 32);
  float rstd = rsqrtf(qs * (1.0f / 128.0f) + LN_EPS);
  float4 g4 = *reinterpret_cast<const float4*>(gamma + d);
  float4 e4 = *reinterpret_cast<const float4*>(beta + d);
  float4 y;
  y.x = ax * rstd * g4.x + e4.x; y.y = ay * rstd * g4.y + e4.y;
  y.z = az * rstd * g4.z + e4.z; y.w = aw * rstd * g4.w + e4.w;
  float* xrow = xf + (size_t)node * D + d;
  if (layer > 0) {                     // residual from previous activations
    float4 r4 = *reinterpret_cast<const float4*>(xrow);
    y.x += r4.x; y.y += r4.y; y.z += r4.z; y.w += r4.w;
  }
  if (layer < NLAYERS - 1) {           // ReLU except after last layer
    y.x = fmaxf(y.x, 0.0f); y.y = fmaxf(y.y, 0.0f);
    y.z = fmaxf(y.z, 0.0f); y.w = fmaxf(y.w, 0.0f);
  }
  *reinterpret_cast<float4*>(xrow) = y;
  union { __bf16 h[4]; uint2 u; } pk;
  pk.h[0] = (__bf16)y.x; pk.h[1] = (__bf16)y.y;
  pk.h[2] = (__bf16)y.z; pk.h[3] = (__bf16)y.w;
  *reinterpret_cast<uint2*>(xb + (size_t)node * D + d) = pk.u;
}

// ---------------- pooling + head ----------------
__global__ __launch_bounds__(256) void k_pool(const float* __restrict__ xf,
                                              const long long* __restrict__ batch,
                                              float* __restrict__ pool, int n) {
  long long gid = (long long)blockIdx.x * blockDim.x + threadIdx.x;
  int node = (int)(gid >> 5);
  if (node >= n) return;
  int q = (int)(gid & 31) << 2;
  long long g = batch[node];
  float4 h4 = *reinterpret_cast<const float4*>(xf + (size_t)node * D + q);
  float* p = pool + g * D + q;
  atomicAdd(p + 0, h4.x); atomicAdd(p + 1, h4.y);
  atomicAdd(p + 2, h4.z); atomicAdd(p + 3, h4.w);
}

__global__ __launch_bounds__(256) void k_cnt(const long long* __restrict__ batch,
                                             float* __restrict__ cnt, int n) {
  int i = blockIdx.x * blockDim.x + threadIdx.x;
  if (i < n) atomicAdd(&cnt[batch[i]], 1.0f);
}

__global__ __launch_bounds__(256) void k_head(const float* __restrict__ pool,
                                              const float* __restrict__ cnt,
                                              const float* __restrict__ lw,
                                              const float* __restrict__ lb,
                                              float* __restrict__ out, int g_total) {
  int wid = threadIdx.x >> 5, lane = threadIdx.x & 31;
  int g = blockIdx.x * 8 + wid;
  if (g >= g_total) return;
  int d = lane << 2;
  float4 s4 = *reinterpret_cast<const float4*>(pool + (size_t)g * D + d);
  float4 w4 = *reinterpret_cast<const float4*>(lw + d);
  float inv = 1.0f / fmaxf(cnt[g], 1.0f);
  float dot = (s4.x * w4.x + s4.y * w4.y + s4.z * w4.z + s4.w * w4.w) * inv;
#pragma unroll
  for (int off = 16; off > 0; off >>= 1) dot += __shfl_xor(dot, off, 32);
  if (lane == 0) out[g] = dot + lb[0];
}

// ---------------- host launcher ----------------
extern "C" void kernel_launch(void* const* d_in, const int* in_sizes, int n_in,
                              void* d_out, int out_size, void* d_ws, size_t ws_size,
                              hipStream_t stream) {
  const float*      x    = (const float*)d_in[0];
  const long long*  ei   = (const long long*)d_in[1];
  const long long*  batch= (const long long*)d_in[2];
  const float*      Ws   = (const float*)d_in[3];
  const float*      bs   = (const float*)d_in[4];
  const float*      gms  = (const float*)d_in[5];
  const float*      bts  = (const float*)d_in[6];
  const float*      lw   = (const float*)d_in[7];
  const float*      lb   = (const float*)d_in[8];
  float*            out  = (float*)d_out;

  const int       N = in_sizes[0] / D;      // 100000
  const long long E = in_sizes[1] / 2;      // 1600000
  const int       G = out_size;             // 512
  const long long src_off = 0, dst_off = E;

  // workspace layout (256B aligned)
  char* w = (char*)d_ws;
  size_t off = 0;
  auto take = [&](size_t bytes) { char* p = w + off; off = (off + bytes + 255) & ~(size_t)255; return p; };
  float*  deg  = (float*) take((size_t)N * 4);
  float*  dis  = (float*) take((size_t)N * 4);
  __bf16* Wt   = (__bf16*)take((size_t)NLAYERS * D * D * 2);
  __bf16* Xb   = (__bf16*)take((size_t)N * D * 2);
  float*  H    = (float*) take((size_t)N * D * 4);
  float*  AGG  = (float*) take((size_t)N * D * 4);
  float*  XF   = (float*) take((size_t)N * D * 4);
  float*  POOL = (float*) take((size_t)G * D * 4);
  float*  CNT  = (float*) take((size_t)G * 4);
  (void)ws_size; (void)n_in;

  const int T = 256;
  auto cdiv = [](long long a, long long b) { return (int)((a + b - 1) / b); };

  // prep: zero accumulators, degree, inv-sqrt-deg, weight transpose+bf16, x->bf16
  k_zero<<<cdiv(N, T), T, 0, stream>>>(deg, N);
  k_zero<<<cdiv((long long)G * D, T), T, 0, stream>>>(POOL, (long long)G * D);
  k_zero<<<cdiv(G, T), T, 0, stream>>>(CNT, G);
  k_deg <<<cdiv(E, T), T, 0, stream>>>(ei + dst_off, deg, E);
  k_dis <<<cdiv(N, T), T, 0, stream>>>(deg, dis, N);
  k_prepw<<<(NLAYERS * D * D) / T, T, 0, stream>>>(Ws, Wt);
  k_cvt <<<cdiv((long long)N * D, T), T, 0, stream>>>(x, Xb, (long long)N * D);

  const int mtiles = (N + 15) / 16;
  const int gblocks = cdiv(mtiles, MT);
  for (int l = 0; l < NLAYERS; ++l) {
    k_gemm<<<gblocks, T, 0, stream>>>(Xb, Wt + (size_t)l * D * D, H, N, mtiles);
    k_agg_init<<<cdiv((long long)N * 32, T), T, 0, stream>>>(H, dis, AGG, N);
    k_scatter <<<cdiv(E * 32, T), T, 0, stream>>>(H, ei + src_off, ei + dst_off, dis, AGG, E);
    k_finalize<<<cdiv(N, 8), T, 0, stream>>>(AGG, bs + l * D, gms + l * D, bts + l * D,
                                             XF, Xb, l, N);
  }

  k_pool<<<cdiv((long long)N * 32, T), T, 0, stream>>>(XF, batch, POOL, N);
  k_cnt <<<cdiv(N, T), T, 0, stream>>>(batch, CNT, N);
  k_head<<<cdiv(G, 8), T, 0, stream>>>(POOL, CNT, lw, lb, out, G);
}
